// CAMPseudoLabel_16793322127809
// MI455X (gfx1250) — matrix operand order repaired and checked by reading, not compiled, not run on hardware
//
#include <hip/hip_runtime.h>

typedef __attribute__((ext_vector_type(16))) _Float16 v16h;
typedef __attribute__((ext_vector_type(8)))  float    v8f;
typedef __attribute__((ext_vector_type(4)))  float    f4;
typedef __attribute__((ext_vector_type(4)))  int      i4;

#define VOXELS (96*96*96)        // 884736
#define NCLS   14
#define GAMMA  0.2f

// ---------------------------------------------------------------------------
// Kernel 1: fused cam / pseudo / updated_masks  (HBM-bound streaming pass,
// ~210MB @ 23.3 TB/s ≈ 9us). 4 voxels per thread via b128 non-temporal
// loads/stores: write-once data, keep it out of L2, quarter the VMEM ops.
// ---------------------------------------------------------------------------
__global__ void cam_fused_kernel(const float* __restrict__ feature,
                                 const int*   __restrict__ masks,
                                 float* __restrict__ cam_out,
                                 float* __restrict__ pseudo_out,
                                 float* __restrict__ umask_out) {
    const long quarter = VOXELS / 4;
    long t = (long)blockIdx.x * blockDim.x + threadIdx.x;
    if (t >= 2 * quarter) return;
    int n = (int)(t / quarter);
    int v = (int)(t % quarter) * 4;
    const float invVol = 1.0f / (float)VOXELS;
    const float* f = feature + (long)n * NCLS * VOXELS + v;

    float best[4]  = {-1.0f, -1.0f, -1.0f, -1.0f};
    int   bestc[4] = {0, 0, 0, 0};
#pragma unroll
    for (int c = 1; c < NCLS; ++c) {
        f4 x = __builtin_nontemporal_load((const f4*)(f + (long)c * VOXELS));
#pragma unroll
        for (int i = 0; i < 4; ++i) {
            float xv = x[i] * invVol;
            xv = fmaxf(xv, 0.0f);              // relu
            xv = (xv > GAMMA) ? xv : 0.0f;     // threshold
            x[i] = xv;
            if (xv > best[i]) { best[i] = xv; bestc[i] = c - 1; }  // first-max
        }
        __builtin_nontemporal_store(x, (f4*)&cam_out[((long)n * 13 + (c - 1)) * VOXELS + v]);
    }
    i4 m = __builtin_nontemporal_load((const i4*)(masks + (long)n * VOXELS + v));
    f4 ps, um;
#pragma unroll
    for (int i = 0; i < 4; ++i) {
        ps[i] = (float)bestc[i];
        um[i] = (float)(m[i] + (m[i] == 0 ? bestc[i] : 0));
    }
    long idx = (long)n * VOXELS + v;
    __builtin_nontemporal_store(ps, (f4*)&pseudo_out[idx]);
    __builtin_nontemporal_store(um, (f4*)&umask_out[idx]);
}

// ---------------------------------------------------------------------------
// f32 -> f16 weight conversion (conv weights become B^T [N,K] for free:
// OIDHW flat index oc*(Cin*27) + c*27 + j  ==  Bt[oc, k] with k = c*27+j)
// ---------------------------------------------------------------------------
__global__ void cvt_f16_kernel(const float* __restrict__ in,
                               _Float16* __restrict__ out, int n) {
    int i = blockIdx.x * blockDim.x + threadIdx.x;
    if (i < n) out[i] = (_Float16)in[i];
}

// ---------------------------------------------------------------------------
// im2col for conv1: dom_feat [2,256,4,4,4], pad 1, 3x3x3
// A1[m, k] : m = n*64 + d*16+h*4+w  (M=128), k = c*27 + kd*9+kh*3+kw (K=6912)
// ---------------------------------------------------------------------------
__global__ void prep1_kernel(const float* __restrict__ x, _Float16* __restrict__ A1) {
    int tid = blockIdx.x * blockDim.x + threadIdx.x;
    const int K = 256 * 27;
    if (tid >= 128 * K) return;
    int m = tid / K, k = tid % K;
    int n = m / 64, sp = m % 64;
    int d = sp / 16, h = (sp / 4) & 3, w = sp & 3;
    int c = k / 27, j = k % 27;
    int dd = d + j / 9 - 1, hh = h + (j / 3) % 3 - 1, ww = w + j % 3 - 1;
    float val = 0.0f;
    if ((unsigned)dd < 4u && (unsigned)hh < 4u && (unsigned)ww < 4u)
        val = x[((long)n * 256 + c) * 64 + dd * 16 + hh * 4 + ww];
    A1[tid] = (_Float16)val;
}

// ---------------------------------------------------------------------------
// Fragment loader (CDNA5 ISA 7.12.2 wave32 layouts):
//  A 16x32 f16: halfs 0..7 -> K = k0 + lanehi*8 + e ; halfs 8..15 -> +16
//  B 32x16 f16: halfs 0..7 -> K = k0 + lanehi*8 + e ; halfs 8..15 -> +16
// ---------------------------------------------------------------------------
__device__ __forceinline__ void load_frags(const _Float16* __restrict__ arow,
                                           const _Float16* __restrict__ brow,
                                           int k0, int lanehi, v16h& a, v16h& b) {
#pragma unroll
    for (int e = 0; e < 8; ++e) {
        a[e]     = arow[k0 + lanehi * 8 + e];
        a[8 + e] = arow[k0 + 16 + lanehi * 8 + e];
    }
#pragma unroll
    for (int e = 0; e < 8; ++e) {
        b[e]     = brow[k0 + lanehi * 8 + e];
        b[8 + e] = brow[k0 + 16 + lanehi * 8 + e];
    }
}

// ---------------------------------------------------------------------------
// K-split, software-pipelined WMMA GEMM.
// Block = kwaves waves; every wave computes the SAME 16x16 tile over a
// disjoint K-slice (steps = K/32/kwaves, exact), partials reduced via LDS.
// Pipelining: fragments for step s+1 are issued before the WMMA of step s,
// so the pre-WMMA wait is "loadcnt <= inflight" instead of 0; plus
// global_prefetch two steps ahead.
// ---------------------------------------------------------------------------
__global__ void wmma_gemm_kernel(const _Float16* __restrict__ A,
                                 const _Float16* __restrict__ Bt,
                                 const float* __restrict__ bias,
                                 float* __restrict__ C,
                                 int M, int N, int K, int kwaves) {
    __shared__ float red[8 * 256];             // up to 8 waves x 16x16 partials

    int lane   = threadIdx.x & 31;
    int wave   = threadIdx.x >> 5;
    int lanelo = lane & 15;
    int lanehi = lane >> 4;

    int tile    = blockIdx.x;
    int tiles_n = N >> 4;
    int tm = (tile / tiles_n) << 4;
    int tn = (tile % tiles_n) << 4;

    int steps = (K >> 5) / kwaves;             // exact by construction
    int kbeg  = wave * steps * 32;

    const _Float16* arow = A  + (long)(tm + lanelo) * K + kbeg;  // row M=lane%16
    const _Float16* brow = Bt + (long)(tn + lanelo) * K + kbeg;  // col N=lane%16

    v8f acc = {};
    v16h a, b, a2, b2;
    load_frags(arow, brow, 0, lanehi, a, b);
    for (int s = 0; s < steps - 1; ++s) {
        if (s + 2 < steps) {
            __builtin_prefetch(arow + (s + 2) * 32, 0, 0);
            __builtin_prefetch(brow + (s + 2) * 32, 0, 0);
        }
        load_frags(arow, brow, (s + 1) * 32, lanehi, a2, b2);   // issue next
        acc = __builtin_amdgcn_wmma_f32_16x16x32_f16(
            false, a, false, b, (short)0, acc, false, false);   // consume cur
        a = a2; b = b2;
    }
    acc = __builtin_amdgcn_wmma_f32_16x16x32_f16(
        false, a, false, b, (short)0, acc, false, false);

    // stash partials: entry e = r*32 + lane  <->  C(tm + r + (l>>4)*8, tn + (l&15))
#pragma unroll
    for (int r = 0; r < 8; ++r)
        red[wave * 256 + r * 32 + lane] = acc[r];
    __syncthreads();

    for (int e = threadIdx.x; e < 256; e += blockDim.x) {
        float s = 0.0f;
        for (int wv = 0; wv < kwaves; ++wv) s += red[wv * 256 + e];
        int l = e & 31, r = e >> 5;
        int cn = tn + (l & 15);
        int cm = tm + r + (l >> 4) * 8;
        C[(long)cm * N + cn] = s + (bias ? bias[cn] : 0.0f);
    }
}

// ---------------------------------------------------------------------------
// relu(maxpool2(conv1)) fused into im2col for conv2.
// C1 is [M=128 rows (n, d*16+h*4+w), N=128 channels].
// A2[m2, k]: m2 = n*8 + pd*4+ph*2+pw (M=16), k = c*27 + j (K=3456), pad 1.
// ---------------------------------------------------------------------------
__global__ void prep2_kernel(const float* __restrict__ C1, _Float16* __restrict__ A2) {
    int tid = blockIdx.x * blockDim.x + threadIdx.x;
    const int K = 128 * 27;
    if (tid >= 16 * K) return;
    int m2 = tid / K, k = tid % K;
    int n = m2 / 8, sp = m2 % 8;
    int pd = sp / 4, ph = (sp / 2) & 1, pw = sp & 1;
    int c = k / 27, j = k % 27;
    int dd = pd + j / 9 - 1, hh = ph + (j / 3) % 3 - 1, ww = pw + j % 3 - 1;
    float val = 0.0f;
    if ((unsigned)dd < 2u && (unsigned)hh < 2u && (unsigned)ww < 2u) {
        float mx = -3.4e38f;
#pragma unroll
        for (int z = 0; z < 2; ++z)
#pragma unroll
            for (int y = 0; y < 2; ++y)
#pragma unroll
                for (int x = 0; x < 2; ++x) {
                    int m = n * 64 + (2 * dd + z) * 16 + (2 * hh + y) * 4 + (2 * ww + x);
                    mx = fmaxf(mx, C1[m * 128 + c]);
                }
        val = fmaxf(mx, 0.0f);   // relu after pool
    }
    A2[tid] = (_Float16)val;
}

// ---------------------------------------------------------------------------
// relu(maxpool2(conv2)) -> [2,64] -> linear(64->1). C2 is [M=16, N=64].
// ---------------------------------------------------------------------------
__global__ void finish_kernel(const float* __restrict__ C2,
                              const float* __restrict__ lin_w,
                              const float* __restrict__ lin_b,
                              float* __restrict__ logits) {
    __shared__ float red[64];
    int c = threadIdx.x, n = blockIdx.x;
    float mx = -3.4e38f;
#pragma unroll
    for (int sp = 0; sp < 8; ++sp)
        mx = fmaxf(mx, C2[(n * 8 + sp) * 64 + c]);
    red[c] = fmaxf(mx, 0.0f) * lin_w[c];
    __syncthreads();
    for (int s = 32; s > 0; s >>= 1) {
        if (c < s) red[c] += red[c + s];
        __syncthreads();
    }
    if (c == 0) logits[n] = red[0] + lin_b[0];
}

// ---------------------------------------------------------------------------
extern "C" void kernel_launch(void* const* d_in, const int* in_sizes, int n_in,
                              void* d_out, int out_size, void* d_ws, size_t ws_size,
                              hipStream_t stream) {
    const float* feature = (const float*)d_in[0];   // [2,14,96,96,96]
    const int*   masks   = (const int*)  d_in[1];   // [2,1,96,96,96]
    const float* domfeat = (const float*)d_in[2];   // [2,256,4,4,4]
    const float* c1w     = (const float*)d_in[3];   // [128,256,3,3,3]
    const float* c1b     = (const float*)d_in[4];   // [128]
    const float* c2w     = (const float*)d_in[5];   // [64,128,3,3,3]
    const float* c2b     = (const float*)d_in[6];   // [64]
    const float* lw      = (const float*)d_in[7];   // [1,64]
    const float* lb      = (const float*)d_in[8];   // [1]

    float* out = (float*)d_out;
    // output layout: cam | pseudo | updated_masks | dom_logits
    float* cam_out    = out;
    float* pseudo_out = out + 2L * 13 * VOXELS;
    float* umask_out  = pseudo_out + 2L * VOXELS;
    float* logits_out = umask_out + 2L * VOXELS;

    // workspace carve-up (256B aligned)
    char* ws = (char*)d_ws;
    size_t off = 0;
    auto alloc = [&](size_t bytes) { void* p = ws + off; off = (off + bytes + 255) & ~(size_t)255; return p; };
    _Float16* A1 = (_Float16*)alloc(128L * 6912 * 2);   // im2col conv1
    _Float16* B1 = (_Float16*)alloc(128L * 6912 * 2);   // conv1_w f16
    float*    C1 = (float*)   alloc(128L * 128 * 4);    // conv1 out [M,N]
    _Float16* A2 = (_Float16*)alloc(16L * 3456 * 2);    // im2col conv2
    _Float16* B2 = (_Float16*)alloc(64L * 3456 * 2);    // conv2_w f16
    float*    C2 = (float*)   alloc(16L * 64 * 4);      // conv2 out [M,N]
    (void)ws_size;

    // 1) HBM-bound fused pass (dominant cost), 4 voxels/thread
    {
        long total = 2L * VOXELS / 4;
        int blocks = (int)((total + 255) / 256);
        cam_fused_kernel<<<blocks, 256, 0, stream>>>(feature, masks,
                                                     cam_out, pseudo_out, umask_out);
    }
    // 2) weight conversions to f16 (B^T layouts come for free from OIDHW)
    cvt_f16_kernel<<<(128 * 6912 + 255) / 256, 256, 0, stream>>>(c1w, B1, 128 * 6912);
    cvt_f16_kernel<<<(64 * 3456 + 255) / 256, 256, 0, stream>>>(c2w, B2, 64 * 3456);
    // 3) im2col conv1
    prep1_kernel<<<(128 * 6912 + 255) / 256, 256, 0, stream>>>(domfeat, A1);
    // 4) conv1 GEMM: [128x128] = [128x6912]*[6912x128], 8-way K-split
    //    64 tiles x 8 waves = 512 waves, 27 pipelined WMMA steps each
    wmma_gemm_kernel<<<(128 / 16) * (128 / 16), 8 * 32, 0, stream>>>(
        A1, B1, c1b, C1, 128, 128, 6912, 8);
    // 5) relu(maxpool) + im2col conv2
    prep2_kernel<<<(16 * 3456 + 255) / 256, 256, 0, stream>>>(C1, A2);
    // 6) conv2 GEMM: [16x64] = [16x3456]*[3456x64], 4-way K-split (108/4=27)
    wmma_gemm_kernel<<<(16 / 16) * (64 / 16), 4 * 32, 0, stream>>>(
        A2, B2, c2b, C2, 16, 64, 3456, 4);
    // 7) relu(maxpool) + linear -> logits
    finish_kernel<<<2, 64, 0, stream>>>(C2, lw, lb, logits_out);
}
